// EchoStateNetwork_27994596835687
// MI455X (gfx1250) — matrix operand order repaired and checked by reading, not compiled, not run on hardware
//
#include <hip/hip_runtime.h>

// ---------------------------------------------------------------------------
// Echo State Network on gfx1250 (MI455X)
//   T=1024 sequential steps of r_new = 0.5*r + 0.5*tanh(W@r + Win@u)
//   W (4096x4096) converted once to bf16 WMMA-fragment layout (32 MB, L2-resident)
//   Persistent kernel, 128 blocks x 256 thr, atomic grid barrier per step.
//   Mat-vec done with v_wmma_f32_16x16x32_bf16 (column-0 GEMV trick).
//   Inner loop kept branch-free + unroll 8 so the scheduler software-pipelines
//   up to 8 chunk loads in flight per wave (hides L2 latency at 2 waves/SIMD).
// ---------------------------------------------------------------------------

typedef __bf16  v16bf __attribute__((ext_vector_type(16)));
typedef float   v8f   __attribute__((ext_vector_type(8)));

#define R_DIM   4096
#define T_STEPS 1024
#define NBLK    128          // persistent grid size (must be co-resident)
#define ROWS_PER_BLK 32      // 2 row-blocks of 16

// workspace layout (bytes)
#define WS_CNT_OFF   0
#define WS_RBUF_OFF  256                       // 2 * 4096 floats (double buffer)
#define WS_PART_OFF  (WS_RBUF_OFF + 2*R_DIM*4) // 1024 * 128 floats
#define WS_WFRAG_OFF (WS_PART_OFF + T_STEPS*NBLK*4)

// ---------------------------------------------------------------------------
// Init: zero r0 and the barrier counter
// ---------------------------------------------------------------------------
__global__ void esn_init(float* rbuf0, unsigned* cnt) {
    int i = blockIdx.x * 256 + threadIdx.x;
    if (i < R_DIM) rbuf0[i] = 0.0f;
    if (i == 0) *cnt = 0u;
}

// ---------------------------------------------------------------------------
// Convert W (fp32 row-major) -> bf16 WMMA A-fragment layout.
// Fragment element index: idx = ((rb*128 + kc)*32 + lane)*16 + j
//   lane 0-15 : M = rb*16 + lane,    K = kc*32 + ((j<8)?j:j+8)
//   lane16-31 : M = rb*16 + lane-16, K = kc*32 + ((j<8)?j:j+8) + 8
// (per CDNA5 ISA 7.12.2, 16-bit A-matrix 16x32 layout)
// ---------------------------------------------------------------------------
__global__ void esn_convert(const float* __restrict__ W,
                            unsigned short* __restrict__ WfragU) {
    __bf16* Wfrag = reinterpret_cast<__bf16*>(WfragU);
    size_t base = ((size_t)blockIdx.x * 256 + threadIdx.x) * 4;
    #pragma unroll
    for (int e = 0; e < 4; ++e) {
        size_t idx   = base + e;
        int j        = (int)(idx & 15);
        int lane     = (int)((idx >> 4) & 31);
        size_t chunk = idx >> 9;
        int kc       = (int)(chunk & 127);
        int rb       = (int)(chunk >> 7);
        int m        = rb * 16 + (lane & 15);
        int half     = lane >> 4;
        int klocal   = ((j < 8) ? j : j + 8) + half * 8;
        int k        = kc * 32 + klocal;
        Wfrag[idx]   = (__bf16)W[(size_t)m * R_DIM + k];
    }
}

// ---------------------------------------------------------------------------
// Device-wide barrier via monotonically increasing atomic counter
// ---------------------------------------------------------------------------
__device__ __forceinline__ void grid_barrier(unsigned* cnt, unsigned target) {
    __syncthreads();
    if (threadIdx.x == 0) {
        __threadfence();
        atomicAdd(cnt, 1u);
        while (atomicAdd(cnt, 0u) < target) __builtin_amdgcn_s_sleep(1);
        __threadfence();
    }
    __syncthreads();
}

// ---------------------------------------------------------------------------
// Persistent step loop
// ---------------------------------------------------------------------------
__global__ void __launch_bounds__(256, 1)
esn_steps(const float* __restrict__ batch,
          const float* __restrict__ Win,
          const float* __restrict__ Wout,
          const unsigned short* __restrict__ WfragU,
          float* __restrict__ rbuf,        // [2][4096] double buffer
          float* __restrict__ partials,    // [1024][128]
          unsigned* cnt) {
    const __bf16* Wfrag = reinterpret_cast<const __bf16*>(WfragU);

    __shared__ float rs[R_DIM];          // staged reservoir state (16 KB)
    __shared__ float part[2][4][16];     // per row-block, per K-quarter partials

    const int tid  = threadIdx.x;
    const int lane = tid & 31;
    const int wid  = tid >> 5;           // 8 waves
    const int rbl  = wid & 1;            // local 16-row block (0/1)
    const int kq   = wid >> 1;           // K quarter (0..3)
    const int rbG  = blockIdx.x * 2 + rbl;
    const __bf16* wf = Wfrag + (size_t)rbG * (128 * 512);

    // B fragment: every column N holds the r chunk, K in table order
    // (K = 8*(lane>>4) + (j<8 ? j : j+8)), matching A.  Only column 0 is used.
    v16bf b;
    #pragma unroll
    for (int j = 0; j < 16; ++j) b[j] = (__bf16)0.0f;

    for (int step = 0; step < T_STEPS; ++step) {
        const int cur = step & 1, nxt = cur ^ 1;

        // stage full r into LDS (b128 loads/stores)
        {
            const float4* src = (const float4*)(rbuf + (size_t)cur * R_DIM);
            float4* dst = (float4*)rs;
            #pragma unroll
            for (int i = tid; i < R_DIM / 4; i += 256) dst[i] = src[i];
        }
        __syncthreads();

        // ---- W @ r for 16 rows, this wave's K quarter, via WMMA bf16 ----
        // Branch-free body: unroll 8 lets the scheduler keep many chunk
        // loads in flight (staggered s_wait_loadcnt) to hide L2 latency.
        v8f acc = {};
        #pragma unroll 8
        for (int i = 0; i < 32; ++i) {
            const int kc = (kq << 5) + i;              // chunk within row-block
            const v16bf* ap = (const v16bf*)(wf + (size_t)kc * 512);
            v16bf a = ap[lane];                        // 32B contiguous per lane
            const int kb = kc * 32 + ((lane >> 4) << 3);
            #pragma unroll
            for (int j = 0; j < 16; ++j)
                b[j] = (__bf16)rs[kb + ((j < 8) ? j : j + 8)]; // column 0 = r chunk
            acc = __builtin_amdgcn_wmma_f32_16x16x32_bf16(
                      false, a, false, b, (short)0, acc, false, false);
        }

        // column 0 of C: lane0 has M=0..7 (vgpr v), lane16 has M=8..15
        if (lane == 0) {
            #pragma unroll
            for (int v = 0; v < 8; ++v) part[rbl][kq][v] = acc[v];
        } else if (lane == 16) {
            #pragma unroll
            for (int v = 0; v < 8; ++v) part[rbl][kq][8 + v] = acc[v];
        }
        __syncthreads();

        // ---- state update for this block's 32 rows (one wave) ----
        if (tid < ROWS_PER_BLK) {
            const int rb2 = tid >> 4, lm = tid & 15;
            float Wr = part[rb2][0][lm] + part[rb2][1][lm]
                     + part[rb2][2][lm] + part[rb2][3][lm];
            const int grow = blockIdx.x * ROWS_PER_BLK + tid;
            const float u  = batch[step];
            const float winu = Win[grow * 2] + Win[grow * 2 + 1] * u;
            const float x  = tanhf(Wr + winu);
            const float rn = 0.5f * (rs[grow] + x);    // leak = 0.5
            rbuf[(size_t)nxt * R_DIM + grow] = rn;
            float pw = Wout[2 + grow] * rn;            // Wout partial
            #pragma unroll
            for (int off = 16; off > 0; off >>= 1)
                pw += __shfl_down(pw, off, 32);
            if (tid == 0)
                partials[(size_t)step * NBLK + blockIdx.x] = pw;
        }

        grid_barrier(cnt, (unsigned)NBLK * (unsigned)(step + 1));
    }
}

// ---------------------------------------------------------------------------
// Fold per-block partials into predictions
// ---------------------------------------------------------------------------
__global__ void esn_out(const float* __restrict__ batch,
                        const float* __restrict__ Wout,
                        const float* __restrict__ partials,
                        float* __restrict__ out) {
    int t = blockIdx.x * 256 + threadIdx.x;
    if (t >= T_STEPS) return;
    float s = Wout[0] + Wout[1] * batch[t];
    const float* p = partials + (size_t)t * NBLK;
    #pragma unroll 8
    for (int b = 0; b < NBLK; ++b) s += p[b];
    out[t] = s;
}

// ---------------------------------------------------------------------------
extern "C" void kernel_launch(void* const* d_in, const int* in_sizes, int n_in,
                              void* d_out, int out_size, void* d_ws, size_t ws_size,
                              hipStream_t stream) {
    const float* batch = (const float*)d_in[0];   // (1024,1,1)
    const float* Win   = (const float*)d_in[1];   // (4096,2)
    const float* W     = (const float*)d_in[2];   // (4096,4096)
    const float* Wout  = (const float*)d_in[3];   // (1,4098)
    float* out = (float*)d_out;

    char* ws = (char*)d_ws;
    unsigned*       cnt      = (unsigned*)(ws + WS_CNT_OFF);
    float*          rbuf     = (float*)(ws + WS_RBUF_OFF);
    float*          partials = (float*)(ws + WS_PART_OFF);
    unsigned short* Wfrag    = (unsigned short*)(ws + WS_WFRAG_OFF);

    esn_init<<<16, 256, 0, stream>>>(rbuf, cnt);
    esn_convert<<<(R_DIM * R_DIM) / (256 * 4), 256, 0, stream>>>(W, Wfrag);
    esn_steps<<<NBLK, 256, 0, stream>>>(batch, Win, Wout, Wfrag,
                                        rbuf, partials, cnt);
    esn_out<<<(T_STEPS + 255) / 256, 256, 0, stream>>>(batch, Wout, partials, out);
}
